// BingramLanguageModel_47906065220079
// MI455X (gfx1250) — compile-verified, matrix operand order checked by reading, not run on hardware
//
#include <hip/hip_runtime.h>
#include <hip/hip_bf16.h>

#define TLEN  2048
#define CDIM  256
#define VOCAB 32000
#define BSZ   4
#define NROWS (BSZ * TLEN)   // 8192

typedef __attribute__((ext_vector_type(16))) __bf16 v16bf;
typedef __attribute__((ext_vector_type(8)))  float  v8f;

union BF16Frag {
    v16bf v;
    unsigned int u[8];
    unsigned short h[16];
};

static __device__ __forceinline__ unsigned short f2bf(float f) {
    unsigned int u = __float_as_uint(f);
    unsigned int r = u + 0x7FFFu + ((u >> 16) & 1u);   // RNE
    return (unsigned short)(r >> 16);
}

static __device__ __forceinline__ v8f wmma_bf16(v16bf a, v16bf b, v8f c) {
    return __builtin_amdgcn_wmma_f32_16x16x32_bf16(
        /*neg_a=*/false, a, /*neg_b=*/false, b,
        /*c_mod=*/(short)0, c, /*reuse_a=*/false, /*reuse_b=*/false);
}

// Load an A-style fragment (lanes index row, K packed in pairs) from a
// row-major [rows, ld] bf16(ushort) matrix with K contiguous.
// Per ISA 16-bit A layout: lane<16 -> K {2v,2v+1} (v<4), {16+2(v-4),...} (v>=4);
// lane>=16 adds +8 to K. Same gather serves B-matrix when B[k,n] = W[n,k].
static __device__ __forceinline__ v16bf load_frag_rowK(
    const unsigned short* p, int ld, int row, int k0, int lane)
{
    const int lm = lane & 15, hf = lane >> 4;
    const unsigned short* r = p + (size_t)(row + lm) * ld + k0 + 8 * hf;
    BF16Frag f;
#pragma unroll
    for (int i = 0; i < 8; ++i) {
        int k = 2 * (i & 3) + 16 * (i >> 2);
        f.u[i] = *(const unsigned int*)(r + k);
    }
    return f.v;
}

// Load a B fragment from a row-major [K, ld] matrix (K is the row index,
// lanes index the column n). Strided in K (used for the V chunk in LDS).
static __device__ __forceinline__ v16bf load_fragB_colN(
    const unsigned short* p, int ld, int col, int lane)
{
    const int lm = lane & 15, hf = lane >> 4;
    const unsigned short* c = p + col + lm;
    BF16Frag f;
#pragma unroll
    for (int i = 0; i < 8; ++i) {
        int k = 8 * hf + 2 * (i & 3) + 16 * (i >> 2);
        f.h[2 * i]     = c[(size_t)k * ld];
        f.h[2 * i + 1] = c[(size_t)(k + 1) * ld];
    }
    return f.v;
}

// Async DMA of 32 rows x 256 bf16 (16 KB) global -> LDS, no VGPR round-trip.
// 32 lanes x 16 B per instruction, 32 instructions; tracked by ASYNCcnt.
static __device__ __forceinline__ void async_copy_v_chunk(
    const unsigned short* g, unsigned short* lds, int lane)
{
    unsigned lbase = (unsigned)(size_t)lds + (unsigned)lane * 16u;
    const char* gbase = (const char*)g + (size_t)lane * 16u;
#pragma unroll 4
    for (int j = 0; j < 32; ++j) {
        unsigned loff = lbase + (unsigned)(j * 512);
        const char* gp = gbase + (size_t)(j * 512);
        asm volatile("global_load_async_to_lds_b128 %0, %1, off"
                     :: "v"(loff), "v"(gp) : "memory");
    }
}

#define WAIT_ASYNC() asm volatile("s_wait_asynccnt 0x0" ::: "memory")
#define WAIT_DS()    asm volatile("s_wait_dscnt 0x0"    ::: "memory")

// ---------------- elementwise kernels ----------------

__global__ void castbf_kernel(const float* __restrict__ in,
                              unsigned short* __restrict__ out, int n)
{
    int i = blockIdx.x * blockDim.x + threadIdx.x;
    if (i < n) out[i] = f2bf(in[i]);
}

__global__ void embed_kernel(const int* __restrict__ idx,
                             const float* __restrict__ tok,
                             const float* __restrict__ pos,
                             unsigned short* __restrict__ x)
{
    int rc = blockIdx.x;       // 0..NROWS-1
    int c  = threadIdx.x;      // 0..CDIM-1
    int t  = rc & (TLEN - 1);
    int token = idx[rc];
    float v = tok[(size_t)token * CDIM + c] + pos[(size_t)t * CDIM + c];
    x[(size_t)rc * CDIM + c] = f2bf(v);
}

// ---------------- QKV projection: [8192,256] x [256,256]^T, 16x64 per wave ----------------

__global__ __launch_bounds__(32) void qkv_gemm_kernel(
    const unsigned short* __restrict__ x,
    const unsigned short* __restrict__ wq,
    const unsigned short* __restrict__ wk,
    const unsigned short* __restrict__ wv,
    unsigned short* __restrict__ q,
    unsigned short* __restrict__ k,
    unsigned short* __restrict__ v)
{
    int lane = threadIdx.x;
    int mt = blockIdx.x;              // 0..511
    int ng = blockIdx.y;              // 0..3  (groups of 64 output cols)
    int which = blockIdx.z;           // 0..2
    const unsigned short* w = (which == 0) ? wq : (which == 1) ? wk : wv;
    unsigned short* out     = (which == 0) ? q  : (which == 1) ? k  : v;

    v8f acc[4] = {{}, {}, {}, {}};
#pragma unroll
    for (int ks = 0; ks < CDIM; ks += 32) {
        v16bf a = load_frag_rowK(x, CDIM, mt * 16, ks, lane);
#pragma unroll
        for (int j = 0; j < 4; ++j) {
            v16bf b = load_frag_rowK(w, CDIM, (ng * 4 + j) * 16, ks, lane);
            acc[j] = wmma_bf16(a, b, acc[j]);
        }
    }
    int lm = lane & 15, hf = lane >> 4;
#pragma unroll
    for (int j = 0; j < 4; ++j)
#pragma unroll
        for (int r = 0; r < 8; ++r) {
            int row = mt * 16 + hf * 8 + r;
            out[(size_t)row * CDIM + (ng * 4 + j) * 16 + lm] = f2bf(acc[j][r]);
        }
}

// ---------------- causal attention, flash-style, one wave per 16-query tile ----------------

__global__ __launch_bounds__(32) void attention_kernel(
    const unsigned short* __restrict__ qb,
    const unsigned short* __restrict__ kb,
    const unsigned short* __restrict__ vb,
    unsigned short* __restrict__ ob)
{
    __shared__ unsigned short vstage[2][32 * CDIM]; // 2 x 16 KB: V chunks (double buffer)
    __shared__ unsigned short pstage[16 * 32];      // 1 KB : P tile (16 x 32) as A-matrix

    int lane = threadIdx.x;
    int qt = blockIdx.x;               // 0..127 query tile
    int b  = blockIdx.y;               // 0..3 batch
    size_t base = (size_t)b * TLEN * CDIM;
    int lm = lane & 15, hf = lane >> 4;

    v16bf qf[8];
#pragma unroll
    for (int ks = 0; ks < 8; ++ks)
        qf[ks] = load_frag_rowK(qb + base, CDIM, qt * 16, ks * 32, lane);

    v8f o[16];
#pragma unroll
    for (int i = 0; i < 16; ++i) o[i] = (v8f){};
    float mrow[8], lrow[8];
#pragma unroll
    for (int r = 0; r < 8; ++r) { mrow[r] = -1e30f; lrow[r] = 0.f; }

    const int nchunk = qt / 2 + 1;     // 32-key chunks up to (incl.) diagonal

    // prologue: kick off async DMA of V chunk 0 while we compute S for chunk 0
    async_copy_v_chunk(vb + base, &vstage[0][0], lane);

    for (int ch = 0; ch < nchunk; ++ch) {
        int s0 = ch * 32;
        int buf = ch & 1;

        // S = q . k^T  (two 16-wide key tiles) -- overlaps the in-flight V DMA
        v8f S0 = {}, S1 = {};
#pragma unroll
        for (int ks = 0; ks < 8; ++ks) {
            v16bf b0 = load_frag_rowK(kb + base, CDIM, s0, ks * 32, lane);
            S0 = wmma_bf16(qf[ks], b0, S0);
        }
#pragma unroll
        for (int ks = 0; ks < 8; ++ks) {
            v16bf b1 = load_frag_rowK(kb + base, CDIM, s0 + 16, ks * 32, lane);
            S1 = wmma_bf16(qf[ks], b1, S1);
        }

        WAIT_ASYNC();          // V chunk `ch` landed in vstage[buf]
        WAIT_DS();             // drain LDS reads of the other buffer
        __syncthreads();

        // start DMA for the next chunk into the other buffer
        if (ch + 1 < nchunk)
            async_copy_v_chunk(vb + base + (size_t)(s0 + 32) * CDIM,
                               &vstage[buf ^ 1][0], lane);

        const float scale = 0.0625f;   // 256^-0.5
        float corr[8];
#pragma unroll
        for (int r = 0; r < 8; ++r) {
            int trow = qt * 16 + hf * 8 + r;
            float s0v = S0[r] * scale, s1v = S1[r] * scale;
            if (s0 + lm > trow)      s0v = -1e30f;   // causal mask
            if (s0 + 16 + lm > trow) s1v = -1e30f;
            float tmax = fmaxf(s0v, s1v);
#pragma unroll
            for (int m_ = 1; m_ < 16; m_ <<= 1)
                tmax = fmaxf(tmax, __shfl_xor(tmax, m_, 32));
            float mnew = fmaxf(mrow[r], tmax);
            float p0 = __expf(s0v - mnew), p1 = __expf(s1v - mnew);
            float rs = p0 + p1;
#pragma unroll
            for (int m_ = 1; m_ < 16; m_ <<= 1)
                rs += __shfl_xor(rs, m_, 32);
            float c_ = __expf(mrow[r] - mnew);
            lrow[r] = lrow[r] * c_ + rs;
            mrow[r] = mnew;
            corr[r] = c_;
            pstage[(hf * 8 + r) * 32 + lm]      = f2bf(p0);
            pstage[(hf * 8 + r) * 32 + 16 + lm] = f2bf(p1);
        }
        // rescale running output by exp(m_old - m_new)
#pragma unroll
        for (int ntile = 0; ntile < 16; ++ntile)
#pragma unroll
            for (int r = 0; r < 8; ++r) o[ntile][r] *= corr[r];

        __syncthreads();   // pstage written
        v16bf pa = load_frag_rowK(pstage, 32, 0, 0, lane);   // P as 16x32 A-matrix
#pragma unroll
        for (int ntile = 0; ntile < 16; ++ntile) {
            v16bf bv = load_fragB_colN(&vstage[buf][0], CDIM, ntile * 16, lane);
            o[ntile] = wmma_bf16(pa, bv, o[ntile]);
        }
        __syncthreads();   // protect pstage before next chunk
    }

    // normalize and store
#pragma unroll
    for (int ntile = 0; ntile < 16; ++ntile)
#pragma unroll
        for (int r = 0; r < 8; ++r) {
            int row = qt * 16 + hf * 8 + r;
            ob[base + (size_t)row * CDIM + ntile * 16 + lm] =
                f2bf(o[ntile][r] / lrow[r]);
        }
}

// ---------------- LM head: [8192,256] x [32000,256]^T + bias, 16x64 per wave ----------------

__global__ __launch_bounds__(32) void lmhead_kernel(
    const unsigned short* __restrict__ ao,
    const unsigned short* __restrict__ wl,
    const float* __restrict__ bl,
    float* __restrict__ logits)
{
    int lane = threadIdx.x;
    int ng = blockIdx.x;   // 0..499 (groups of 64 vocab cols)
    int mt = blockIdx.y;   // 0..511

    v8f acc[4] = {{}, {}, {}, {}};
#pragma unroll
    for (int ks = 0; ks < CDIM; ks += 32) {
        v16bf a = load_frag_rowK(ao, CDIM, mt * 16, ks, lane);
#pragma unroll
        for (int j = 0; j < 4; ++j) {
            v16bf b = load_frag_rowK(wl, CDIM, (ng * 4 + j) * 16, ks, lane);
            acc[j] = wmma_bf16(a, b, acc[j]);
        }
    }
    int lm = lane & 15, hf = lane >> 4;
#pragma unroll
    for (int j = 0; j < 4; ++j) {
        int col = (ng * 4 + j) * 16 + lm;
        float bias = bl[col];
#pragma unroll
        for (int r = 0; r < 8; ++r) {
            int row = mt * 16 + hf * 8 + r;
            logits[(size_t)row * VOCAB + col] = acc[j][r] + bias;
        }
    }
}

// ---------------- cross-entropy ----------------

__global__ __launch_bounds__(256) void rowloss_kernel(
    const float* __restrict__ logits,
    const int* __restrict__ tgt,
    float* __restrict__ rowloss)
{
    __shared__ float red[256];
    int m = blockIdx.x;
    int t = threadIdx.x;
    const float* row = logits + (size_t)m * VOCAB;

    float mx = -1e30f;
    for (int i = t; i < VOCAB; i += 256) mx = fmaxf(mx, row[i]);
    red[t] = mx; __syncthreads();
    for (int s = 128; s > 0; s >>= 1) {
        if (t < s) red[t] = fmaxf(red[t], red[t + s]);
        __syncthreads();
    }
    float rmax = red[0];
    __syncthreads();

    float se = 0.f;
    for (int i = t; i < VOCAB; i += 256) se += __expf(row[i] - rmax);
    red[t] = se; __syncthreads();
    for (int s = 128; s > 0; s >>= 1) {
        if (t < s) red[t] += red[t + s];
        __syncthreads();
    }
    if (t == 0) {
        float lse = rmax + __logf(red[0]);
        rowloss[m] = lse - row[tgt[m]];
    }
}

__global__ __launch_bounds__(256) void finalize_loss_kernel(
    const float* __restrict__ rowloss, float* __restrict__ out)
{
    __shared__ float red[256];
    int t = threadIdx.x;
    float s = 0.f;
    for (int i = t; i < NROWS; i += 256) s += rowloss[i];
    red[t] = s; __syncthreads();
    for (int k = 128; k > 0; k >>= 1) {
        if (t < k) red[t] += red[t + k];
        __syncthreads();
    }
    if (t == 0) out[0] = red[0] / (float)NROWS;
}

// ---------------- launch ----------------

extern "C" void kernel_launch(void* const* d_in, const int* in_sizes, int n_in,
                              void* d_out, int out_size, void* d_ws, size_t ws_size,
                              hipStream_t stream)
{
    const int*   idx = (const int*)d_in[0];
    const int*   tgt = (const int*)d_in[1];
    const float* tok = (const float*)d_in[2];
    const float* pos = (const float*)d_in[3];
    const float* Wq  = (const float*)d_in[4];
    const float* Wk  = (const float*)d_in[5];
    const float* Wv  = (const float*)d_in[6];
    const float* Wl  = (const float*)d_in[7];
    const float* bl  = (const float*)d_in[8];

    float* logits = (float*)d_out;
    float* loss   = logits + (size_t)NROWS * VOCAB;

    unsigned short* ws  = (unsigned short*)d_ws;
    unsigned short* xbf = ws;                                   // 8192*256
    unsigned short* qbf = xbf + (size_t)NROWS * CDIM;
    unsigned short* kbf = qbf + (size_t)NROWS * CDIM;
    unsigned short* vbf = kbf + (size_t)NROWS * CDIM;
    unsigned short* obf = vbf + (size_t)NROWS * CDIM;
    unsigned short* wqb = obf + (size_t)NROWS * CDIM;
    unsigned short* wkb = wqb + (size_t)CDIM * CDIM;
    unsigned short* wvb = wkb + (size_t)CDIM * CDIM;
    unsigned short* wlb = wvb + (size_t)CDIM * CDIM;
    float* rowloss = (float*)(wlb + (size_t)VOCAB * CDIM);      // 8192 floats

    // weights -> bf16
    castbf_kernel<<<(CDIM * CDIM + 255) / 256, 256, 0, stream>>>(Wq, wqb, CDIM * CDIM);
    castbf_kernel<<<(CDIM * CDIM + 255) / 256, 256, 0, stream>>>(Wk, wkb, CDIM * CDIM);
    castbf_kernel<<<(CDIM * CDIM + 255) / 256, 256, 0, stream>>>(Wv, wvb, CDIM * CDIM);
    castbf_kernel<<<(VOCAB * CDIM + 255) / 256, 256, 0, stream>>>(Wl, wlb, VOCAB * CDIM);

    // x = tok[idx] + pos
    embed_kernel<<<NROWS, CDIM, 0, stream>>>(idx, tok, pos, xbf);

    // q,k,v
    qkv_gemm_kernel<<<dim3(NROWS / 16, CDIM / 64, 3), 32, 0, stream>>>(
        xbf, wqb, wkb, wvb, qbf, kbf, vbf);

    // causal attention
    attention_kernel<<<dim3(TLEN / 16, BSZ), 32, 0, stream>>>(qbf, kbf, vbf, obf);

    // logits
    lmhead_kernel<<<dim3(VOCAB / 64, NROWS / 16), 32, 0, stream>>>(obf, wlb, bl, logits);

    // cross-entropy
    rowloss_kernel<<<NROWS, 256, 0, stream>>>(logits, tgt, rowloss);
    finalize_loss_kernel<<<1, 256, 0, stream>>>(rowloss, loss);
}
